// CrystalGraphConvNet_67937792688479
// MI455X (gfx1250) — matrix operand order repaired and checked by reading, not compiled, not run on hardware
//
#include <hip/hip_runtime.h>

// ---------------- problem constants ----------------
#define N_ATOMS 16384
#define M_NBR   12
#define OA      92
#define OB      41
#define AF      128
#define BF      64
#define LDIM    2
#define NC      3
#define BATCH   128
#define APB     (N_ATOMS / BATCH)        // atoms per crystal = 128
#define RROWS   (N_ATOMS * M_NBR)        // 196608 rows of the conv GEMM
#define KDIM    (2 * AF + BF)            // 320
#define KT_     (KDIM / 32)              // 10 k-tiles of 32
#define RT_     (RROWS / 16)             // 12288 row-tiles of 16
#define PDIM    (AF + M_NBR * BF)        // 896 pooled dim
#define TIN_C   (2 * OA + OB)            // 225
#define ZDEC_ELEMS ((size_t)RROWS * TIN_C)   // 44,236,800

#define TPB 256
#define CEILDIV(a, b) (((a) + (b) - 1) / (b))

typedef __attribute__((ext_vector_type(16))) __bf16 v16bf;
typedef __attribute__((ext_vector_type(8)))  float  v8f;

// ---------------- device helpers ----------------
__device__ __forceinline__ float softplusf(float x) {
  return (x > 20.0f) ? x : log1pf(__expf(x));
}
__device__ __forceinline__ float sigmoidf(float x) {
  return 1.0f / (1.0f + __expf(-x));
}
__device__ __forceinline__ unsigned short f2bf(float f) {
  unsigned u = __float_as_uint(f);
  unsigned r = u + 0x7FFFu + ((u >> 16) & 1u);   // round-to-nearest-even
  return (unsigned short)(r >> 16);
}
__device__ __forceinline__ float gauss_hash(unsigned i) {
  unsigned s = i * 747796405u + 2891336453u;
  s ^= s >> 17; s *= 0xed5ad4bbu; s ^= s >> 11; s *= 0xac4c1b51u;
  s ^= s >> 15; s *= 0x31848babu; s ^= s >> 14;
  unsigned a = s;
  unsigned b = s * 0x9E3779B9u + 0x7F4A7C15u;
  b ^= b >> 16; b *= 0x85ebca6bu; b ^= b >> 13;
  float u1 = (float)(a >> 8) * (1.0f / 16777216.0f) + 1e-7f;
  float u2 = (float)(b >> 8) * (1.0f / 16777216.0f);
  return sqrtf(-2.0f * __logf(u1)) * __cosf(6.2831853f * u2);
}

// ---------------- misc ----------------
__global__ void fill_zero(float* __restrict__ p, int n) {
  int t = blockIdx.x * blockDim.x + threadIdx.x;
  if (t < n) p[t] = 0.0f;
}

// ---------------- embeddings ----------------
__global__ __launch_bounds__(TPB) void embed_atom_kernel(
    const float* __restrict__ X, const float* __restrict__ W,
    const float* __restrict__ bia, float* __restrict__ A) {
  int t = blockIdx.x * blockDim.x + threadIdx.x;
  if (t >= N_ATOMS * AF) return;
  int n = t >> 7, f = t & (AF - 1);
  float s = bia[f];
  const float* x = X + (size_t)n * OA;
  for (int k = 0; k < OA; ++k) s += x[k] * W[k * AF + f];
  A[t] = s;
}

__global__ __launch_bounds__(TPB) void embed_bond_kernel(
    const float* __restrict__ X, const float* __restrict__ W,
    const float* __restrict__ bia, float* __restrict__ NB) {
  int t = blockIdx.x * blockDim.x + threadIdx.x;
  if (t >= RROWS * BF) return;
  int nm = t >> 6, f = t & (BF - 1);
  float s = bia[f];
  const float* x = X + (size_t)nm * OB;
  for (int k = 0; k < OB; ++k) s += x[k] * W[k * BF + f];
  NB[t] = s;
}

// total_input_fea output: [atom | atom[idx] | nbr] on ORIGINAL features
__global__ __launch_bounds__(TPB) void write_total_input(
    const float* __restrict__ atom, const float* __restrict__ nbr,
    const int* __restrict__ idx, float* __restrict__ out) {
  long long t = (long long)blockIdx.x * blockDim.x + threadIdx.x;
  if (t >= (long long)RROWS * TIN_C) return;
  int nm = (int)(t / TIN_C), c = (int)(t % TIN_C);
  int n = nm / M_NBR;
  float v;
  if (c < OA)            v = atom[(size_t)n * OA + c];
  else if (c < 2 * OA)   v = atom[(size_t)idx[nm] * OA + (c - OA)];
  else                   v = nbr[(size_t)nm * OB + (c - 2 * OA)];
  out[t] = v;
}

// ---------------- WMMA GEMM path ----------------
// Pack A = [a | a[idx] | nb] into bf16 16x32 A-fragment order:
//   offset = ((rt*KT + kt)*32 + lane)*16 ; lane row = rt*16 + lane%16
//   k (e<8)  = kt*32 + (lane<16 ? 0 : 8) + e
//   k (e>=8) = kt*32 + 16 + (lane<16 ? 0 : 8) + (e-8)
__global__ __launch_bounds__(TPB) void build_total_pack(
    const float* __restrict__ A, const float* __restrict__ NB,
    const int* __restrict__ idx, unsigned short* __restrict__ Ap) {
  int t = blockIdx.x * blockDim.x + threadIdx.x;
  if (t >= RT_ * KT_ * 32) return;
  int lane = t & 31;
  int kt = (t >> 5) % KT_;
  int rt = (t >> 5) / KT_;
  int nm = rt * 16 + (lane & 15);
  int n = nm / M_NBR;
  int nj = idx[nm];
  int kb = (lane < 16) ? 0 : 8;
  unsigned short o[16];
#pragma unroll
  for (int e = 0; e < 16; ++e) {
    int k = kt * 32 + ((e < 8) ? (kb + e) : (16 + kb + e - 8));
    float v;
    if (k < AF)            v = A[(size_t)n * AF + k];
    else if (k < 2 * AF)   v = A[(size_t)nj * AF + (k - AF)];
    else                   v = NB[(size_t)nm * BF + (k - 2 * AF)];
    o[e] = f2bf(v);
  }
  unsigned w[8];
#pragma unroll
  for (int i = 0; i < 8; ++i)
    w[i] = (unsigned)o[2 * i] | ((unsigned)o[2 * i + 1] << 16);
  uint4* dst = reinterpret_cast<uint4*>(Ap + (size_t)t * 16);
  dst[0] = make_uint4(w[0], w[1], w[2], w[3]);
  dst[1] = make_uint4(w[4], w[5], w[6], w[7]);
}

// Pack W (KDIM x NOUT, row-major f32) into bf16 32x16 B-fragment order:
//   offset = ((kt*NT + nt)*32 + lane)*16 ; col = nt*16 + lane%16
//   k (e<8) = kt*32 + (lane<16 ? 0 : 8) + e ; k (e>=8) = kt*32 + 16 + (lane<16 ? 0 : 8) + (e-8)
__global__ __launch_bounds__(TPB) void pack_weights(
    const float* __restrict__ W, unsigned short* __restrict__ Wp,
    int NT, int ldw) {
  int t = blockIdx.x * blockDim.x + threadIdx.x;
  if (t >= KT_ * NT * 32) return;
  int lane = t & 31;
  int nt = (t >> 5) % NT;
  int kt = (t >> 5) / NT;
  int col = nt * 16 + (lane & 15);
  int k0 = kt * 32 + ((lane < 16) ? 0 : 8);
  unsigned short o[16];
#pragma unroll
  for (int e = 0; e < 16; ++e) o[e] = f2bf(W[(size_t)(k0 + ((e < 8) ? e : e + 8)) * ldw + col]);
  unsigned w[8];
#pragma unroll
  for (int i = 0; i < 8; ++i)
    w[i] = (unsigned)o[2 * i] | ((unsigned)o[2 * i + 1] << 16);
  uint4* dst = reinterpret_cast<uint4*>(Wp + (size_t)t * 16);
  dst[0] = make_uint4(w[0], w[1], w[2], w[3]);
  dst[1] = make_uint4(w[4], w[5], w[6], w[7]);
}

// G(RROWS x NT*16) = Apacked @ Wpacked.
// The entire packed weight block (KT_*NT*512 bf16) is staged into LDS once per
// workgroup via async global->LDS copies (ASYNCcnt), then each wave computes a
// 16x64 strip (4 accumulators) with B fragments served from LDS.
__global__ __launch_bounds__(TPB) void gemm_bf16_wmma(
    const unsigned short* __restrict__ Ap, const unsigned short* __restrict__ Wp,
    float* __restrict__ G, int NT) {
  extern __shared__ unsigned short shW[];

  // ---- cooperative async stage of the whole weight block into LDS ----
  {
    const int nchunk = (KT_ * NT * 512) / 8;          // 16-byte chunks
    unsigned lbase = (unsigned)(unsigned long long)&shW[0];  // LDS byte address
    for (int c = threadIdx.x; c < nchunk; c += TPB) {
      unsigned ldsoff = lbase + (unsigned)c * 16u;
      const unsigned short* g = Wp + (size_t)c * 8;
      asm volatile("global_load_async_to_lds_b128 %0, %1, off"
                   :: "v"(ldsoff), "v"(g) : "memory");
    }
    asm volatile("s_wait_asynccnt 0x0" ::: "memory");
  }
  __syncthreads();

  const int wave = (blockIdx.x * blockDim.x + threadIdx.x) >> 5;
  const int lane = threadIdx.x & 31;
  const int strips = NT >> 2;
  const int rt = wave / strips;
  if (rt < RT_) {
    const int nt0 = (wave - rt * strips) << 2;
    v8f c0 = {}, c1 = {}, c2 = {}, c3 = {};
    const unsigned short* ap = Ap + ((size_t)rt * KT_ * 32 + lane) * 16;
    const unsigned short* wb = shW + ((size_t)nt0 * 32 + lane) * 16;
    const size_t wstride = (size_t)NT * 512;
    for (int kt = 0; kt < KT_; ++kt) {
      v16bf av = *(const v16bf*)ap;
      const unsigned short* wp = wb + (size_t)kt * wstride;
      v16bf b0 = *(const v16bf*)(wp);
      v16bf b1 = *(const v16bf*)(wp + 512);
      v16bf b2 = *(const v16bf*)(wp + 1024);
      v16bf b3 = *(const v16bf*)(wp + 1536);
      c0 = __builtin_amdgcn_wmma_f32_16x16x32_bf16(false, av, false, b0, (short)0, c0, false, false);
      c1 = __builtin_amdgcn_wmma_f32_16x16x32_bf16(false, av, false, b1, (short)0, c1, false, false);
      c2 = __builtin_amdgcn_wmma_f32_16x16x32_bf16(false, av, false, b2, (short)0, c2, false, false);
      c3 = __builtin_amdgcn_wmma_f32_16x16x32_bf16(false, av, false, b3, (short)0, c3, false, false);
      ap += 512;
    }
    const int col = lane & 15;
    const int rbase = (lane >> 4) << 3;   // 0 or 8
    const int ldg = NT << 4;
    float* gb = G + ((size_t)rt * 16 + rbase) * ldg + (nt0 << 4) + col;
#pragma unroll
    for (int r = 0; r < 8; ++r) {
      float* gr = gb + (size_t)r * ldg;
      gr[0] = c0[r]; gr[16] = c1[r]; gr[32] = c2[r]; gr[48] = c3[r];
    }
  }
}

// ---------------- BatchNorm reductions ----------------
__global__ __launch_bounds__(TPB) void bn_stats(
    const float* __restrict__ X, float* __restrict__ sum, float* __restrict__ sq,
    int R, int C, int rowsPerBlock) {
  int r0 = blockIdx.x * rowsPerBlock;
  int r1 = r0 + rowsPerBlock; if (r1 > R) r1 = R;
  for (int c = threadIdx.x; c < C; c += blockDim.x) {
    float s = 0.0f, q = 0.0f;
    for (int r = r0; r < r1; ++r) {
      float x = X[(size_t)r * C + c];
      s += x; q += x * x;
    }
    atomicAdd(&sum[c], s);
    atomicAdd(&sq[c], q);
  }
}

__global__ void bn_finalize(const float* __restrict__ sum, const float* __restrict__ sq,
                            float* __restrict__ mean, float* __restrict__ inv,
                            int C, float invR) {
  int c = blockIdx.x * blockDim.x + threadIdx.x;
  if (c >= C) return;
  float m = sum[c] * invR;
  float v = sq[c] * invR - m * m;
  mean[c] = m;
  inv[c] = rsqrtf(v + 1e-5f);
}

// ---------------- conv epilogues ----------------
__global__ __launch_bounds__(TPB) void conv_asum(
    const float* __restrict__ G, const float* __restrict__ mean, const float* __restrict__ inv,
    const float* __restrict__ g1a, const float* __restrict__ b1a, float* __restrict__ asum) {
  int t = blockIdx.x * blockDim.x + threadIdx.x;
  if (t >= N_ATOMS * AF) return;
  int n = t >> 7, f = t & (AF - 1);
  float mF = mean[f],      iF = inv[f],      gF = g1a[f],      bF = b1a[f];
  float mC = mean[f + AF], iC = inv[f + AF], gC = g1a[f + AF], bC = b1a[f + AF];
  float s = 0.0f;
  for (int m = 0; m < M_NBR; ++m) {
    size_t base = (size_t)(n * M_NBR + m) * (2 * AF);
    float xf = (G[base + f]      - mF) * iF * gF + bF;
    float xc = (G[base + AF + f] - mC) * iC * gC + bC;
    s += sigmoidf(xf) * softplusf(xc);
  }
  asum[t] = s;
}

__global__ __launch_bounds__(TPB) void update_a(
    float* __restrict__ A, const float* __restrict__ asum,
    const float* __restrict__ mean2, const float* __restrict__ inv2,
    const float* __restrict__ g2a, const float* __restrict__ b2a) {
  int t = blockIdx.x * blockDim.x + threadIdx.x;
  if (t >= N_ATOMS * AF) return;
  int f = t & (AF - 1);
  float bn = (asum[t] - mean2[f]) * inv2[f] * g2a[f] + b2a[f];
  A[t] = softplusf(A[t] + bn);
}

__global__ __launch_bounds__(TPB) void update_nb(
    float* __restrict__ NB, const float* __restrict__ G,
    const float* __restrict__ mean, const float* __restrict__ inv,
    const float* __restrict__ g1b, const float* __restrict__ b1b) {
  int t = blockIdx.x * blockDim.x + threadIdx.x;
  if (t >= RROWS * BF) return;
  int nm = t >> 6, j = t & (BF - 1);
  size_t base = (size_t)nm * (2 * BF);
  float xf = (G[base + j]      - mean[j])      * inv[j]      * g1b[j]      + b1b[j];
  float xc = (G[base + BF + j] - mean[BF + j]) * inv[BF + j] * g1b[BF + j] + b1b[BF + j];
  NB[t] = softplusf(NB[t] + sigmoidf(xf) * softplusf(xc));
}

// ---------------- VAE middle ----------------
__global__ __launch_bounds__(TPB) void pool_kernel(
    const float* __restrict__ A, const float* __restrict__ NB, float* __restrict__ pooled) {
  int t = blockIdx.x * blockDim.x + threadIdx.x;
  if (t >= BATCH * PDIM) return;
  int b = t / PDIM, c = t % PDIM;
  float s = 0.0f;
  for (int i = 0; i < APB; ++i) {
    int n = b * APB + i;
    s += (c < AF) ? A[(size_t)n * AF + c] : NB[(size_t)n * (M_NBR * BF) + (c - AF)];
  }
  pooled[t] = softplusf(s / (float)APB);
}

__global__ void mu_logvar_kernel(
    const float* __restrict__ pooled,
    const float* __restrict__ muW, const float* __restrict__ muB,
    const float* __restrict__ lvW, const float* __restrict__ lvB,
    float* __restrict__ muWS, float* __restrict__ lvWS,
    float* __restrict__ outMu, float* __restrict__ outLv) {
  int t = blockIdx.x * blockDim.x + threadIdx.x;
  if (t >= BATCH * LDIM) return;
  int b = t / LDIM, l = t % LDIM;
  float sm = muB[l], sl = lvB[l];
  const float* p = pooled + (size_t)b * PDIM;
  for (int c = 0; c < PDIM; ++c) {
    float pv = p[c];
    sm += pv * muW[c * LDIM + l];
    sl += pv * lvW[c * LDIM + l];
  }
  muWS[t] = sm; lvWS[t] = sl;
  outMu[t] = sm; outLv[t] = sl;
}

__global__ void z_kernel(const float* __restrict__ muWS, const float* __restrict__ lvWS,
                         float* __restrict__ zWS, float* __restrict__ outZ) {
  int t = blockIdx.x * blockDim.x + threadIdx.x;
  if (t >= BATCH * LDIM) return;
  float z = muWS[t] + gauss_hash((unsigned)t) * __expf(0.5f * lvWS[t]);
  zWS[t] = z; outZ[t] = z;
}

__global__ __launch_bounds__(TPB) void zd_kernel(
    const float* __restrict__ zWS, const float* __restrict__ decW,
    const float* __restrict__ decB, float* __restrict__ zd) {
  int t = blockIdx.x * blockDim.x + threadIdx.x;
  if (t >= BATCH * PDIM) return;
  int b = t / PDIM, c = t % PDIM;
  zd[t] = zWS[b * LDIM] * decW[c] + zWS[b * LDIM + 1] * decW[PDIM + c] + decB[c];
}

__global__ __launch_bounds__(TPB) void dec_kernel(
    float* __restrict__ A, float* __restrict__ NB,
    const float* __restrict__ zd, const float* __restrict__ pooled) {
  long long t = (long long)blockIdx.x * blockDim.x + threadIdx.x;
  if (t >= (long long)N_ATOMS * PDIM) return;
  int n = (int)(t / PDIM), c = (int)(t % PDIM);
  int b = n / APB;
  float np = (c < AF) ? A[(size_t)n * AF + c] : NB[(size_t)n * (M_NBR * BF) + (c - AF)];
  float v = softplusf(np * zd[(size_t)b * PDIM + c] / pooled[(size_t)b * PDIM + c]);
  if (c < AF) A[(size_t)n * AF + c] = v;
  else        NB[(size_t)n * (M_NBR * BF) + (c - AF)] = v;
}

// ---------------- final embeddings / assembly ----------------
__global__ __launch_bounds__(TPB) void final_atom_kernel(
    const float* __restrict__ A, const float* __restrict__ W,
    const float* __restrict__ bia, float* __restrict__ ZA) {
  int t = blockIdx.x * blockDim.x + threadIdx.x;
  if (t >= N_ATOMS * OA) return;
  int n = t / OA, o = t % OA;
  float s = bia[o];
  const float* a = A + (size_t)n * AF;
  for (int k = 0; k < AF; ++k) s += a[k] * W[k * OA + o];
  ZA[t] = sigmoidf(s);
}

__global__ __launch_bounds__(TPB) void final_bond_kernel(
    const float* __restrict__ NB, const float* __restrict__ W,
    const float* __restrict__ bia, float* __restrict__ zdec) {
  int t = blockIdx.x * blockDim.x + threadIdx.x;
  if (t >= RROWS * OB) return;
  int nm = t / OB, o = t % OB;
  float s = bia[o];
  const float* x = NB + (size_t)nm * BF;
  for (int k = 0; k < BF; ++k) s += x[k] * W[k * OB + o];
  zdec[(size_t)nm * TIN_C + 2 * OA + o] = sigmoidf(s);
}

__global__ __launch_bounds__(TPB) void assemble_kernel(
    const float* __restrict__ ZA, const int* __restrict__ idx, float* __restrict__ zdec) {
  int t = blockIdx.x * blockDim.x + threadIdx.x;
  if (t >= RROWS * OA) return;
  int nm = t / OA, o = t % OA;
  int n = nm / M_NBR;
  zdec[(size_t)nm * TIN_C + o]      = ZA[(size_t)n * OA + o];
  zdec[(size_t)nm * TIN_C + OA + o] = ZA[(size_t)idx[nm] * OA + o];
}

// ---------------- host: one conv layer ----------------
static void run_conv(hipStream_t stream,
                     const float* Wa, const float* Wb,
                     const float* g1a, const float* b1a,
                     const float* g1b, const float* b1b,
                     const float* g2a, const float* b2a,
                     const int* idx, float* A, float* NB,
                     unsigned short* Ap, unsigned short* Wp,
                     float* G, float* asum, float* stats) {
  // stats layout (512-float regions): sum, sq, mean, inv, sum2, sq2, mean2, inv2
  float *SUM = stats, *SQ = stats + 512, *MEAN = stats + 1024, *INV = stats + 1536;
  float *SUM2 = stats + 2048, *SQ2 = stats + 2560, *MEAN2 = stats + 3072, *INV2 = stats + 3584;

  pack_weights<<<CEILDIV(KT_ * 16 * 32, TPB), TPB, 0, stream>>>(Wa, Wp, 16, 2 * AF);
  build_total_pack<<<CEILDIV(RT_ * KT_ * 32, TPB), TPB, 0, stream>>>(A, NB, idx, Ap);
  gemm_bf16_wmma<<<CEILDIV(RT_ * 4 * 32, TPB), TPB, KT_ * 16 * 512 * 2, stream>>>(Ap, Wp, G, 16);

  fill_zero<<<CEILDIV(1024, TPB), TPB, 0, stream>>>(SUM, 1024);
  bn_stats<<<RROWS / 512, TPB, 0, stream>>>(G, SUM, SQ, RROWS, 2 * AF, 512);
  bn_finalize<<<1, 256, 0, stream>>>(SUM, SQ, MEAN, INV, 2 * AF, 1.0f / (float)RROWS);

  conv_asum<<<CEILDIV(N_ATOMS * AF, TPB), TPB, 0, stream>>>(G, MEAN, INV, g1a, b1a, asum);
  fill_zero<<<CEILDIV(1024, TPB), TPB, 0, stream>>>(SUM2, 1024);
  bn_stats<<<N_ATOMS / 512, TPB, 0, stream>>>(asum, SUM2, SQ2, N_ATOMS, AF, 512);
  bn_finalize<<<1, 128, 0, stream>>>(SUM2, SQ2, MEAN2, INV2, AF, 1.0f / (float)N_ATOMS);
  update_a<<<CEILDIV(N_ATOMS * AF, TPB), TPB, 0, stream>>>(A, asum, MEAN2, INV2, g2a, b2a);

  pack_weights<<<CEILDIV(KT_ * 8 * 32, TPB), TPB, 0, stream>>>(Wb, Wp, 8, 2 * BF);
  gemm_bf16_wmma<<<CEILDIV(RT_ * 2 * 32, TPB), TPB, KT_ * 8 * 512 * 2, stream>>>(Ap, Wp, G, 8);

  fill_zero<<<CEILDIV(1024, TPB), TPB, 0, stream>>>(SUM, 1024);
  bn_stats<<<RROWS / 512, TPB, 0, stream>>>(G, SUM, SQ, RROWS, 2 * BF, 512);
  bn_finalize<<<1, 128, 0, stream>>>(SUM, SQ, MEAN, INV, 2 * BF, 1.0f / (float)RROWS);
  update_nb<<<CEILDIV(RROWS * BF, TPB), TPB, 0, stream>>>(NB, G, MEAN, INV, g1b, b1b);
}

// ---------------- entry ----------------
extern "C" void kernel_launch(void* const* d_in, const int* in_sizes, int n_in,
                              void* d_out, int out_size, void* d_ws, size_t ws_size,
                              hipStream_t stream) {
  (void)in_sizes; (void)n_in; (void)out_size; (void)ws_size;
  const float* atom_fea  = (const float*)d_in[0];
  const float* nbr_fea   = (const float*)d_in[1];
  const int*   idx       = (const int*)d_in[2];
  const float* embAW = (const float*)d_in[4];  const float* embAb = (const float*)d_in[5];
  const float* embBW = (const float*)d_in[6];  const float* embBb = (const float*)d_in[7];
  const float* embA2W = (const float*)d_in[8]; const float* embA2b = (const float*)d_in[9];
  const float* embB2W = (const float*)d_in[10];const float* embB2b = (const float*)d_in[11];
  const float* muW = (const float*)d_in[12];   const float* muB = (const float*)d_in[13];
  const float* lvW = (const float*)d_in[14];   const float* lvB = (const float*)d_in[15];
  const float* decW = (const float*)d_in[16];  const float* decB = (const float*)d_in[17];
  const float* c1Wa = (const float*)d_in[18];  const float* c1Wb = (const float*)d_in[19];
  const float* c1g1a = (const float*)d_in[20]; const float* c1b1a = (const float*)d_in[21];
  const float* c1g1b = (const float*)d_in[22]; const float* c1b1b = (const float*)d_in[23];
  const float* c1g2a = (const float*)d_in[24]; const float* c1b2a = (const float*)d_in[25];
  const float* c2Wa = (const float*)d_in[26];  const float* c2Wb = (const float*)d_in[27];
  const float* c2g1a = (const float*)d_in[28]; const float* c2b1a = (const float*)d_in[29];
  const float* c2g1b = (const float*)d_in[30]; const float* c2b1b = (const float*)d_in[31];
  const float* c2g2a = (const float*)d_in[32]; const float* c2b2a = (const float*)d_in[33];

  float* out = (float*)d_out;
  const size_t MU_OFF = ZDEC_ELEMS;
  const size_t LV_OFF = MU_OFF + BATCH * LDIM;
  const size_t Z_OFF  = LV_OFF + BATCH * LDIM;
  const size_t TIN_OFF = Z_OFF + BATCH * LDIM;

  // workspace carve-out (256B aligned)
  char* ws = (char*)d_ws;
  size_t off = 0;
  auto carve = [&](size_t bytes) -> char* {
    char* p = ws + off;
    off = (off + bytes + 255) & ~(size_t)255;
    return p;
  };
  float* A      = (float*)carve((size_t)N_ATOMS * AF * 4);            // 8 MB
  float* NB     = (float*)carve((size_t)RROWS * BF * 4);              // 48 MB
  float* ASUM   = (float*)carve((size_t)N_ATOMS * AF * 4);            // 8 MB
  float* ZA     = (float*)carve((size_t)N_ATOMS * OA * 4);            // 6 MB
  float* POOLED = (float*)carve((size_t)BATCH * PDIM * 4);
  float* ZD     = (float*)carve((size_t)BATCH * PDIM * 4);
  float* MUWS   = (float*)carve((size_t)BATCH * LDIM * 4);
  float* LVWS   = (float*)carve((size_t)BATCH * LDIM * 4);
  float* ZWS    = (float*)carve((size_t)BATCH * LDIM * 4);
  float* STATS  = (float*)carve(4096 * 4);
  unsigned short* WP = (unsigned short*)carve((size_t)KT_ * 16 * 512 * 2);      // 160 KB
  unsigned short* AP = (unsigned short*)carve((size_t)RT_ * KT_ * 512 * 2);     // 126 MB
  float* G      = (float*)carve((size_t)RROWS * 2 * AF * 4);                    // 201 MB

  // encode + output #5
  embed_atom_kernel<<<CEILDIV(N_ATOMS * AF, TPB), TPB, 0, stream>>>(atom_fea, embAW, embAb, A);
  embed_bond_kernel<<<CEILDIV(RROWS * BF, TPB), TPB, 0, stream>>>(nbr_fea, embBW, embBb, NB);
  write_total_input<<<(int)CEILDIV((size_t)RROWS * TIN_C, TPB), TPB, 0, stream>>>(
      atom_fea, nbr_fea, idx, out + TIN_OFF);

  for (int i = 0; i < NC; ++i)
    run_conv(stream, c1Wa + (size_t)i * KDIM * 2 * AF, c1Wb + (size_t)i * KDIM * 2 * BF,
             c1g1a + i * 2 * AF, c1b1a + i * 2 * AF, c1g1b + i * 2 * BF, c1b1b + i * 2 * BF,
             c1g2a + i * AF, c1b2a + i * AF, idx, A, NB, AP, WP, G, ASUM, STATS);

  pool_kernel<<<CEILDIV(BATCH * PDIM, TPB), TPB, 0, stream>>>(A, NB, POOLED);
  mu_logvar_kernel<<<1, TPB, 0, stream>>>(POOLED, muW, muB, lvW, lvB, MUWS, LVWS,
                                          out + MU_OFF, out + LV_OFF);
  z_kernel<<<1, TPB, 0, stream>>>(MUWS, LVWS, ZWS, out + Z_OFF);
  zd_kernel<<<CEILDIV(BATCH * PDIM, TPB), TPB, 0, stream>>>(ZWS, decW, decB, ZD);
  dec_kernel<<<(int)CEILDIV((size_t)N_ATOMS * PDIM, TPB), TPB, 0, stream>>>(A, NB, ZD, POOLED);

  for (int i = 0; i < NC; ++i)
    run_conv(stream, c2Wa + (size_t)i * KDIM * 2 * AF, c2Wb + (size_t)i * KDIM * 2 * BF,
             c2g1a + i * 2 * AF, c2b1a + i * 2 * AF, c2g1b + i * 2 * BF, c2b1b + i * 2 * BF,
             c2g2a + i * AF, c2b2a + i * AF, idx, A, NB, AP, WP, G, ASUM, STATS);

  final_atom_kernel<<<CEILDIV(N_ATOMS * OA, TPB), TPB, 0, stream>>>(A, embA2W, embA2b, ZA);
  final_bond_kernel<<<CEILDIV(RROWS * OB, TPB), TPB, 0, stream>>>(NB, embB2W, embB2b, out);
  assemble_kernel<<<CEILDIV(RROWS * OA, TPB), TPB, 0, stream>>>(ZA, idx, out);
}
